// Fusion_MLP_12429635354826
// MI455X (gfx1250) — compile-verified, matrix operand order checked by reading, not run on hardware
//
#include <hip/hip_runtime.h>
#include <cstdint>

// ---------------------------------------------------------------------------
// CDNA5 (gfx1250) transformer block: cross-attn + MLP, wave32 + WMMA.
//
// All GEMM operands are f16-resident in global memory, stored PRE-PERMUTED
// into the WMMA 16-bit fragment K-order (swap of K bits 3<->4). LDS staging
// is a pure identity copy of 16B blocks via the CDNA5 async-DMA path
// (GLOBAL_LOAD_ASYNC_TO_LDS_B128, ASYNCcnt), double-buffered so the DMA of
// tile k+1 overlaps the WMMAs of tile k; ONE workgroup barrier per K-step.
// Fragments are contiguous 32B LDS reads (2 x ds_load_b128 per fragment).
// Math: v_wmma_f32_16x16x32_f16 (f16 A/B, f32 accumulate).
//
// 256-thread block = 8 waves (4M x 2N); macro tile 64x64; K step 32.
// All shapes are exact multiples of the tiles: no bounds checks in the GEMM.
// ---------------------------------------------------------------------------

typedef __attribute__((ext_vector_type(16))) _Float16 v16h;
typedef __attribute__((ext_vector_type(8)))  _Float16 v8h;
typedef __attribute__((ext_vector_type(8)))  float    v8f;
typedef __attribute__((ext_vector_type(4)))  float    v4f;

#define GBM 64
#define GBN 64
#define GKS 32
#define LDT 40                 // padded LDS row stride in halves (80B, 16B-aligned)
#define BOFF (GBM * LDT)       // Bs base (halves) within one buffer
#define BUFH (2 * BOFF)        // halves per double-buffer slot (As+Bs)

union FragU { v16h v; v8h h[2]; };

// elementwise perm of k in [0,32): swap bits 3 and 4 (fragment K-order)
__device__ __forceinline__ int permElt(int k) {
    return (k & 7) | ((k & 8) << 1) | ((k & 16) >> 1);
}
// perm applied to the low 5 bits of a flat index (rows are 32-aligned)
__device__ __forceinline__ long long permFlat(long long i) {
    return (i & ~31LL) | (long long)permElt((int)(i & 31));
}

// CDNA5 async copy: 16 bytes global -> LDS, tracked by ASYNCcnt.
// NOTE: the LDS operand is derived from the real shared-memory pointer
// (ptrtoint escape) so the compiler treats this asm as a writer of SMEM —
// otherwise it proves SMEM store-free and folds the ds_loads to undef.
// Low 32 bits of a generic LDS pointer are the LDS byte address (ISA §10.2).
__device__ __forceinline__ void async_ld_b128(void* lds, const void* gaddr) {
    asm volatile("global_load_async_to_lds_b128 %0, %1, off"
                 :: "v"((unsigned)(uintptr_t)lds),
                    "v"((unsigned long long)(uintptr_t)gaddr)
                 : "memory");
}
__device__ __forceinline__ void wait_async0() {
    asm volatile("s_wait_asynccnt 0x0" ::: "memory");
}

// Batched GEMM: C = [relu](A*B + bias) + res, with optional f16 side-output.
//   A : f16 [M,K] row-major, K-permuted       (+ sA per batch)
//   BT: f16 [N,ldbt] row-major, K-permuted    (+ sB per batch); B(k,n)=BT[n][k]
//   C : optional f32 [M,N]                    (+ sC per batch)
//   Ch: optional f16 copy of C, fragment-ready:
//       chMode==1 -> Ch[m*N + perm(n)]   (row-major, K-permuted for next GEMM)
//       chMode==2 -> Ch[n*M + perm(m)]   (transposed, K-permuted: BT for next)
__global__ __launch_bounds__(256)
void gemm_wmma_f16(const _Float16* __restrict__ A, const _Float16* __restrict__ BT,
                   const float* __restrict__ bias, const float* __restrict__ res,
                   float* __restrict__ C, _Float16* __restrict__ Ch,
                   int M, int N, int K, int ldbt,
                   long long sA, long long sB, long long sC, long long sRes,
                   int relu, int chMode)
{
    const int bz = blockIdx.z;
    A  += (long long)bz * sA;
    BT += (long long)bz * sB;
    if (C)   C   += (long long)bz * sC;
    if (res) res += (long long)bz * sRes;

    __shared__ _Float16 SMEM[2 * BUFH];   // ping-pong: [buf][As | Bs]

    const int tid  = threadIdx.x;
    const int wave = tid >> 5;
    const int lane = tid & 31;
    const int wm   = wave & 3;           // 0..3 : M subtile
    const int wn   = wave >> 2;          // 0..1 : N pair of subtiles
    const int lm   = lane & 15;
    const int hi   = lane >> 4;

    const int m0 = blockIdx.y * GBM;
    const int n0 = blockIdx.x * GBN;

    // staging decomposition: thread -> (row 0..63, 8-half block 0/8/16/24)
    const int arow = tid >> 2;
    const int ablk = (tid & 3) * 8;

    const _Float16* agBase = A  + (long long)(m0 + arow) * K    + ablk;
    const _Float16* bgBase = BT + (long long)(n0 + arow) * ldbt + ablk;

    auto stage = [&](int buf, int k0) {
        _Float16* sb = SMEM + buf * BUFH;
        async_ld_b128(&sb[arow * LDT + ablk], agBase + k0);
        async_ld_b128(&sb[BOFF + arow * LDT + ablk], bgBase + k0);
    };

    v8f acc0 = {}, acc1 = {};
    stage(0, 0);
    int cur = 0;

    for (int k0 = 0; k0 < K; k0 += GKS) {
        wait_async0();         // our async writes (incl. buffer `cur`) done
        __syncthreads();       // all waves' writes done; prev reads retired
        if (k0 + GKS < K) stage(cur ^ 1, k0 + GKS);   // overlap DMA w/ WMMA

        const _Float16* sb = SMEM + cur * BUFH;
        FragU af, bf0, bf1;
        const int aoff = (wm * 16 + lm) * LDT + hi * 16;
        af.h[0] = *(const v8h*)&sb[aoff];
        af.h[1] = *(const v8h*)&sb[aoff + 8];
        const int boff0 = BOFF + (wn * 32 + lm) * LDT + hi * 16;
        bf0.h[0] = *(const v8h*)&sb[boff0];
        bf0.h[1] = *(const v8h*)&sb[boff0 + 8];
        const int boff1 = boff0 + 16 * LDT;
        bf1.h[0] = *(const v8h*)&sb[boff1];
        bf1.h[1] = *(const v8h*)&sb[boff1 + 8];

        acc0 = __builtin_amdgcn_wmma_f32_16x16x32_f16(
                   false, af.v, false, bf0.v, (short)0, acc0, false, false);
        acc1 = __builtin_amdgcn_wmma_f32_16x16x32_f16(
                   false, af.v, false, bf1.v, (short)0, acc1, false, false);
        cur ^= 1;
    }

    // epilogue: f32 C/D layout (VGPR r: lanes 0-15 -> M=r, 16-31 -> M=8+r)
#pragma unroll
    for (int t = 0; t < 2; ++t) {
        const v8f acc = t ? acc1 : acc0;
        const int nn = n0 + wn * 32 + t * 16 + lm;
        const float bv = bias ? bias[nn] : 0.0f;
        const int pn = (nn & ~31) | permElt(nn & 31);
#pragma unroll
        for (int r = 0; r < 8; ++r) {
            const int mm = m0 + wm * 16 + (hi ? (8 + r) : r);
            float v = acc[r] + bv;
            if (relu) v = v > 0.0f ? v : 0.0f;
            if (res)  v += res[(long long)mm * N + nn];
            if (C)    C[(long long)mm * N + nn] = v;
            if (Ch) {
                if (chMode == 1)
                    Ch[(long long)mm * N + pn] = (_Float16)v;
                else
                    Ch[(long long)nn * M + ((mm & ~31) | permElt(mm & 31))] = (_Float16)v;
            }
        }
    }
}

// f32 -> f16 with fragment K-permutation of the low 5 index bits.
// Requires n % 256 == 0 and row length % 32 == 0 (true for all uses).
__global__ __launch_bounds__(256)
void cvt_perm_f16(const float* __restrict__ src, _Float16* __restrict__ dst)
{
    const long long i = (long long)blockIdx.x * 256 + threadIdx.x;
    dst[permFlat(i)] = (_Float16)src[i];
}

// Weight convert: src f32 [K,N] -> dst f16 [N,K] transposed + K-permuted.
__global__ __launch_bounds__(256)
void cvt_perm_trans_f16(const float* __restrict__ src, _Float16* __restrict__ dst,
                        int K, int N)
{
    const long long i = (long long)blockIdx.x * 256 + threadIdx.x;
    const int k = (int)(i / N), n = (int)(i % N);
    dst[(long long)n * K + ((k & ~31) | permElt(k & 31))] = (_Float16)src[i];
}

// In-place row softmax (pre-scale) + fragment-ready f16 copy.
__global__ __launch_bounds__(256)
void softmax_rows(float* __restrict__ s, _Float16* __restrict__ sh,
                  int L, float scale)
{
    float* p = s + (long long)blockIdx.x * L;
    _Float16* ph = sh + (long long)blockIdx.x * L;
    __shared__ float red[256];
    const int t = threadIdx.x;

    float mx = -3.402823466e38f;
    for (int i = t; i < L; i += 256) mx = fmaxf(mx, p[i]);
    red[t] = mx;
    __syncthreads();
    for (int o = 128; o > 0; o >>= 1) {
        if (t < o) red[t] = fmaxf(red[t], red[t + o]);
        __syncthreads();
    }
    mx = red[0] * scale;
    __syncthreads();

    float sum = 0.0f;
    for (int i = t; i < L; i += 256) {
        float e = __expf(p[i] * scale - mx);
        p[i] = e;
        sum += e;
    }
    red[t] = sum;
    __syncthreads();
    for (int o = 128; o > 0; o >>= 1) {
        if (t < o) red[t] += red[t + o];
        __syncthreads();
    }
    const float inv = 1.0f / red[0];
    for (int i = t; i < L; i += 256) {
        float e = p[i] * inv;
        p[i] = e;
        ph[permFlat((long long)i)] = (_Float16)e;
    }
}

// LayerNorm over one flattened row (L=245760, %1024==0) per block.
// float4 loads; f64 stats; optional fragment-ready f16 copy.
__global__ __launch_bounds__(256)
void layernorm_rows(const float* __restrict__ x, const float* __restrict__ g,
                    const float* __restrict__ b, float* __restrict__ y,
                    _Float16* __restrict__ yh, int L)
{
    const long long base = (long long)blockIdx.x * L;
    const v4f* xr = (const v4f*)(x + base);
    const v4f* gv = (const v4f*)g;
    const v4f* bv = (const v4f*)b;
    v4f* yr = (v4f*)(y + base);
    const int L4 = L >> 2;
    const int t = threadIdx.x;

    double s = 0.0, ss = 0.0;
    for (int i = t; i < L4; i += 256) {
        v4f v = xr[i];
        s  += (double)v[0] + (double)v[1] + (double)v[2] + (double)v[3];
        ss += (double)v[0] * v[0] + (double)v[1] * v[1]
            + (double)v[2] * v[2] + (double)v[3] * v[3];
    }
    __shared__ double rs[256], rq[256];
    rs[t] = s; rq[t] = ss;
    __syncthreads();
    for (int o = 128; o > 0; o >>= 1) {
        if (t < o) { rs[t] += rs[t + o]; rq[t] += rq[t + o]; }
        __syncthreads();
    }
    const double mu  = rs[0] / (double)L;
    const double var = rq[0] / (double)L - mu * mu;
    const float  mean = (float)mu;
    const float  inv  = rsqrtf((float)var + 1e-5f);

    for (int i = t; i < L4; i += 256) {
        v4f v = xr[i], gg = gv[i], bb = bv[i], o;
        o[0] = (v[0] - mean) * inv * gg[0] + bb[0];
        o[1] = (v[1] - mean) * inv * gg[1] + bb[1];
        o[2] = (v[2] - mean) * inv * gg[2] + bb[2];
        o[3] = (v[3] - mean) * inv * gg[3] + bb[3];
        yr[i] = o;
        if (yh) {
            const long long j = base + 4LL * i;   // 4-aligned: same 32-chunk
#pragma unroll
            for (int c = 0; c < 4; ++c) yh[permFlat(j + c)] = (_Float16)o[c];
        }
    }
}

// ---------------------------------------------------------------------------
// Host-side orchestration (graph-capture safe: launches only, on `stream`).
// ---------------------------------------------------------------------------
extern "C" void kernel_launch(void* const* d_in, const int* in_sizes, int n_in,
                              void* d_out, int out_size, void* d_ws, size_t ws_size,
                              hipStream_t stream)
{
    (void)in_sizes; (void)n_in; (void)out_size; (void)ws_size;

    const int B = 64, S = 320, D1 = 512, D2 = 768, H = 192;
    const int BS = B * S;                       // 20480
    const int DM = S * D2;                      // 245760
    const long long SZ = (long long)BS * D2;    // 15,728,640

    const float* x1   = (const float*)d_in[0];
    const float* x2   = (const float*)d_in[1];
    const float* Wq   = (const float*)d_in[2];
    const float* Wk   = (const float*)d_in[3];
    const float* Wv   = (const float*)d_in[4];
    const float* g12  = (const float*)d_in[5];
    const float* b12  = (const float*)d_in[6];
    const float* g13  = (const float*)d_in[7];
    const float* b13  = (const float*)d_in[8];
    const float* W11  = (const float*)d_in[9];
    const float* bs11 = (const float*)d_in[10];
    const float* W12  = (const float*)d_in[11];
    const float* bs12 = (const float*)d_in[12];

    float* out_x2   = (float*)d_out;            // [B,S,D2]
    float* out_attn = (float*)d_out + SZ;       // [B,S,S]

    // Workspace layout (byte bump allocator; ~392 MB total).
    char* wp = (char*)d_ws;
    auto ah = [&](long long n) { _Float16* r = (_Float16*)wp; wp += n * 2; return r; };
    auto af = [&](long long n) { float* r = (float*)wp; wp += n * 4; return r; };

    _Float16* x1h  = ah((long long)BS * D1);    // x1, K-permuted
    _Float16* x2h  = ah(SZ);                    // x2, K-permuted
    _Float16* WqT  = ah((long long)D1 * D2);    // [N=768][K=512]
    _Float16* WkT  = ah((long long)D2 * D2);
    _Float16* WvT  = ah((long long)D2 * D2);
    _Float16* W11T = ah((long long)D2 * H);     // [N=192][K=768]
    _Float16* W12T = ah((long long)H * D2);     // [N=768][K=192]
    _Float16* q1h  = ah(SZ);                    // [BS,768] permuted
    _Float16* k2h  = ah(SZ);                    // [BS,768] permuted (BT for QK^T)
    _Float16* v2hT = ah(SZ);                    // [e=768][BS] transposed+permuted
    _Float16* atth = ah((long long)B * S * S);  // attn, permuted (A of attn@V)
    _Float16* xnh  = ah(SZ);                    // LN1 out f16 permuted
    _Float16* hbh  = ah((long long)BS * H);     // FFN hidden f16 permuted
    float*    xa   = af(SZ);                    // x2 + attention out (f32)
    float*    xn   = af(SZ);                    // LN1 out (f32, residual)
    float*    xb   = af(SZ);                    // xn + FFN out (f32)

    const dim3 blk(256);
    auto gg = [](int M, int N, int Z) { return dim3(N / GBN, M / GBM, Z); };
    auto cg = [](long long n) { return dim3((unsigned)(n / 256)); };

    // 0) f32 -> f16 fragment-permuted conversions
    cvt_perm_f16<<<cg((long long)BS * D1), blk, 0, stream>>>(x1, x1h);
    cvt_perm_f16<<<cg(SZ), blk, 0, stream>>>(x2, x2h);
    cvt_perm_trans_f16<<<cg((long long)D1 * D2), blk, 0, stream>>>(Wq, WqT, D1, D2);
    cvt_perm_trans_f16<<<cg((long long)D2 * D2), blk, 0, stream>>>(Wk, WkT, D2, D2);
    cvt_perm_trans_f16<<<cg((long long)D2 * D2), blk, 0, stream>>>(Wv, WvT, D2, D2);
    cvt_perm_trans_f16<<<cg((long long)D2 * H), blk, 0, stream>>>(W11, W11T, D2, H);
    cvt_perm_trans_f16<<<cg((long long)H * D2), blk, 0, stream>>>(W12, W12T, H, D2);

    // 1-3) projections (f16 outputs only; q1/k2 row-permuted, v2 transposed)
    gemm_wmma_f16<<<gg(BS, D2, 1), blk, 0, stream>>>(
        x1h, WqT, nullptr, nullptr, nullptr, q1h,
        BS, D2, D1, D1, 0, 0, 0, 0, 0, /*chMode=*/1);
    gemm_wmma_f16<<<gg(BS, D2, 1), blk, 0, stream>>>(
        x2h, WkT, nullptr, nullptr, nullptr, k2h,
        BS, D2, D2, D2, 0, 0, 0, 0, 0, 1);
    gemm_wmma_f16<<<gg(BS, D2, 1), blk, 0, stream>>>(
        x2h, WvT, nullptr, nullptr, nullptr, v2hT,
        BS, D2, D2, D2, 0, 0, 0, 0, 0, /*chMode=*/2);

    // 4) scores[b] = q1[b] @ k2[b]^T  (f32, into the attn region of d_out)
    gemm_wmma_f16<<<gg(S, S, B), blk, 0, stream>>>(
        q1h, k2h, nullptr, nullptr, out_attn, nullptr,
        S, S, D2, /*ldbt=*/D2,
        (long long)S * D2, (long long)S * D2, (long long)S * S, 0, 0, 0);

    // 5) softmax (scale = 1/sqrt(768)) in place + permuted f16 copy
    const float scale = 1.0f / sqrtf((float)D2);
    softmax_rows<<<dim3(BS), blk, 0, stream>>>(out_attn, atth, S, scale);

    // 6) xa[b] = attn[b] @ v2[b] + x2[b]   (BT = v2hT, row stride BS)
    gemm_wmma_f16<<<gg(S, D2, B), blk, 0, stream>>>(
        atth, v2hT, nullptr, x2, xa, nullptr,
        S, D2, S, /*ldbt=*/BS,
        (long long)S * S, /*sB=*/S, (long long)S * D2, (long long)S * D2, 0, 0);

    // 7) LN1: xn = LN(xa)*g12 + b12  (+ permuted f16 copy)
    layernorm_rows<<<dim3(B), blk, 0, stream>>>(xa, g12, b12, xn, xnh, DM);

    // 8) hb = relu(xn @ W11 + bias11)  (f16 permuted output only)
    gemm_wmma_f16<<<gg(BS, H, 1), blk, 0, stream>>>(
        xnh, W11T, bs11, nullptr, nullptr, hbh,
        BS, H, D2, D2, 0, 0, 0, 0, /*relu=*/1, 1);

    // 9) xb = hb @ W12 + bias12 + xn
    gemm_wmma_f16<<<gg(BS, D2, 1), blk, 0, stream>>>(
        hbh, W12T, bs12, xn, xb, nullptr,
        BS, D2, H, H, 0, 0, 0, 0, 0, 0);

    // 10) LN2 -> final x2 into d_out.
    layernorm_rows<<<dim3(B), blk, 0, stream>>>(xb, g13, b13, out_x2, nullptr, DM);
}
